// CausalAttention_15461882265954
// MI455X (gfx1250) — matrix operand order repaired and checked by reading, not compiled, not run on hardware
//
#include <hip/hip_runtime.h>
#include <hip/hip_bf16.h>

// ---------------------------------------------------------------------------
// CDNA5 (gfx1250) causal/sliding-window attention, bf16 WMMA + fp32 softmax.
// Wave32. GEMMs: one wave computes a 16x64 strip, software-pipelined k-loop
// (next iteration's A/B operands loaded before this iteration's 4 WMMAs so
// s_wait_loadcnt retires early instead of stalling the XDL pipe).
// Attention: one wave owns 32 queries (two 16-row tiles sharing every K/V
// operand); 32 keys per chunk; global_prefetch_b8 for streaming K/V.
// ---------------------------------------------------------------------------

typedef __bf16 bf16;
typedef __attribute__((ext_vector_type(8)))  bf16  v8bf;
typedef __attribute__((ext_vector_type(16))) bf16  v16bf;
typedef __attribute__((ext_vector_type(8)))  float v8f;

#define WMMA_BF16(A, B, C) \
  __builtin_amdgcn_wmma_f32_16x16x32_bf16(false, (A), false, (B), (short)0, (C), false, false)

static __device__ __forceinline__ unsigned short f2bf(float f) {
  unsigned u = __float_as_uint(f);
  u += 0x7FFFu + ((u >> 16) & 1u);   // round-to-nearest-even
  return (unsigned short)(u >> 16);
}
static __device__ __forceinline__ v8bf ld8(const unsigned short* p) {
  return *reinterpret_cast<const v8bf*>(p);
}
static __device__ __forceinline__ v16bf cat16(v8bf a, v8bf b) {
  return __builtin_shufflevector(a, b, 0,1,2,3,4,5,6,7,8,9,10,11,12,13,14,15);
}
static __device__ __forceinline__ v16bf ld16(const unsigned short* p) {
  return cat16(ld8(p), ld8(p + 8));
}
// Row reductions over 16-lane halves (C/D layout: a row lives in one VGPR
// slot across lanes 0-15 / 16-31; xor masks 1,2,4,8 stay within a half).
static __device__ __forceinline__ float redmax16(float v) {
  v = fmaxf(v, __shfl_xor(v, 1, 32));
  v = fmaxf(v, __shfl_xor(v, 2, 32));
  v = fmaxf(v, __shfl_xor(v, 4, 32));
  v = fmaxf(v, __shfl_xor(v, 8, 32));
  return v;
}
static __device__ __forceinline__ float redsum16(float v) {
  v += __shfl_xor(v, 1, 32);
  v += __shfl_xor(v, 2, 32);
  v += __shfl_xor(v, 4, 32);
  v += __shfl_xor(v, 8, 32);
  return v;
}

// ---------------------------------------------------------------------------
// fp32 -> bf16 conversion
// ---------------------------------------------------------------------------
__global__ void cvt_f32_bf16(const float* __restrict__ src,
                             unsigned short* __restrict__ dst, int n) {
  int i = blockIdx.x * blockDim.x + threadIdx.x;
  if (i < n) dst[i] = f2bf(src[i]);
}

// ---------------------------------------------------------------------------
// QKV GEMM: qkv[n][o] = sum_c x[n][c] * w[o][c] + b[o]   (o in [0,1536))
// One wave per 16x64 strip; k-loop software-pipelined (double-buffered
// operands). Scatter into Q[h][n][64], K[h][n][64], Vt[h][64][n].
// A 16x32 bf16 layout: lane = M + 16*(K>>4); elems = 16 contiguous K.
// B 32x16 bf16 layout: lane = N + 16*(K>>4); elems = 16 contiguous K.
// ---------------------------------------------------------------------------
__global__ __launch_bounds__(128) void qkv_gemm(
    const unsigned short* __restrict__ xb, const unsigned short* __restrict__ wb,
    const float* __restrict__ bias,
    unsigned short* __restrict__ Qb, unsigned short* __restrict__ Kb,
    unsigned short* __restrict__ Vt, int N) {
  const int lane = threadIdx.x & 31;
  const int wv   = threadIdx.x >> 5;
  const int ln   = lane & 15, hi = lane >> 4;
  const int TO   = 24;                       // 1536/64 strips
  int tile = blockIdx.x * 4 + wv;
  int tm = tile / TO, to = tile % TO;
  if (tm * 16 >= N) return;

  const unsigned short* arow = xb + (size_t)(tm * 16 + ln) * 512;
  const unsigned short* br[4];
#pragma unroll
  for (int t = 0; t < 4; t++)
    br[t] = wb + (size_t)(to * 64 + 16 * t + ln) * 512;

  // Prologue: load k-step 0 operands.
  v16bf A = ld16(arow + 16 * hi);
  v16bf B[4];
#pragma unroll
  for (int t = 0; t < 4; t++) B[t] = ld16(br[t] + 16 * hi);

  v8f acc[4] = {};
#pragma unroll 2
  for (int k0 = 0; k0 < 512; k0 += 32) {
    // Issue next iteration's loads first (branchless clamp on last iter).
    const int kn = (k0 + 32 < 512) ? (k0 + 32) : k0;
    __builtin_prefetch(arow + k0 + 96, 0, 0);
    v16bf An = ld16(arow + kn + 16 * hi);
    v16bf Bn[4];
#pragma unroll
    for (int t = 0; t < 4; t++) {
      __builtin_prefetch(br[t] + k0 + 96, 0, 0);
      Bn[t] = ld16(br[t] + kn + 16 * hi);
    }
    // Consume current operands: 4 back-to-back WMMAs.
#pragma unroll
    for (int t = 0; t < 4; t++) acc[t] = WMMA_BF16(A, B[t], acc[t]);
    A = An;
#pragma unroll
    for (int t = 0; t < 4; t++) B[t] = Bn[t];
  }

#pragma unroll
  for (int t = 0; t < 4; t++) {
    const int o   = to * 64 + 16 * t + ln;   // seg/head uniform per 16-col tile
    const float bv = bias[o];
    const int seg = o >> 9;                  // 0=Q 1=K 2=V
    const int h   = (o >> 6) & 7;
    const int d   = o & 63;
#pragma unroll
    for (int i = 0; i < 8; i++) {
      int n = tm * 16 + i + 8 * hi;
      unsigned short v = f2bf(acc[t][i] + bv);
      if (seg == 0)      Qb[((size_t)h * N + n) * 64 + d] = v;
      else if (seg == 1) Kb[((size_t)h * N + n) * 64 + d] = v;
      else               Vt[((size_t)h * 64 + d) * N + n] = v;
    }
  }
}

// ---------------------------------------------------------------------------
// Flash attention. One wave per (head, 32-query group): two 16-row tiles A/B
// share all K/V operands. 32 keys per chunk (8 S-WMMA + 8 PV-WMMA).
// Mask semantics (matches reference exactly):
//   k <  s_end                         -> always allowed (even k > q!)
//   k >= s_end: allowed iff k <= q and frame(k) >= frame(q) - w + 1
// ---------------------------------------------------------------------------
__global__ __launch_bounds__(128) void attn_kernel(
    const unsigned short* __restrict__ Qb, const unsigned short* __restrict__ Kb,
    const unsigned short* __restrict__ Vt, unsigned short* __restrict__ OB,
    int N, const int* __restrict__ p_fs, const int* __restrict__ p_sw,
    const int* __restrict__ p_nfpb, const int* __restrict__ p_nffs) {
  __shared__ unsigned short Plds[4][32 * 32];   // per-wave P staging (bf16)
  const int lane = threadIdx.x & 31;
  const int wv   = threadIdx.x >> 5;
  const int ln   = lane & 15, hi = lane >> 4;
  unsigned short* pl = Plds[wv];

  const int qgroups = N >> 5;
  int tid = blockIdx.x * 4 + wv;
  int h = tid / qgroups, qg = tid % qgroups;
  if (h >= 8) return;
  const int q0 = qg * 32;                    // 32 queries; same frame (32|256)

  const int fs    = p_fs[0];
  const int w     = p_sw[0] * p_nfpb[0];
  const int s_end = p_nffs[0] * fs;          // always-allowed key prefix

  // Q tiles A (rows q0..q0+15) and B (rows q0+16..q0+31), A-layout, 2 d-chunks.
  const unsigned short* qrA = Qb + ((size_t)h * N + q0 + ln) * 64;
  const unsigned short* qrB = qrA + (size_t)16 * 64;
  const v16bf qA0 = ld16(qrA +  0 + 16 * hi);
  const v16bf qA1 = ld16(qrA + 32 + 16 * hi);
  const v16bf qB0 = ld16(qrB +  0 + 16 * hi);
  const v16bf qB1 = ld16(qrB + 32 + 16 * hi);

  v8f oA[4] = {}, oB[4] = {};
  float mA[8], lA[8], mB[8], lB[8];
#pragma unroll
  for (int i = 0; i < 8; i++) {
    mA[i] = mB[i] = -3.0e38f; lA[i] = lB[i] = 0.0f;
  }

  auto chunk = [&](int kb, bool do_mask) {
    // K as B-operand: lane = key(ln) + 16*(K>>4); 16 contiguous d per lane.
    const unsigned short* kr0 = Kb + ((size_t)h * N + kb + ln) * 64;
    const unsigned short* kr1 = kr0 + (size_t)16 * 64;
    __builtin_prefetch(kr0 + (size_t)32 * 64, 0, 0);   // next chunk's K rows
    __builtin_prefetch(kr1 + (size_t)32 * 64, 0, 0);
    const v16bf k00 = ld16(kr0 +  0 + 16 * hi);
    const v16bf k01 = ld16(kr0 + 32 + 16 * hi);
    const v16bf k10 = ld16(kr1 +  0 + 16 * hi);
    const v16bf k11 = ld16(kr1 + 32 + 16 * hi);

    v8f sA0 = {}, sA1 = {}, sB0 = {}, sB1 = {};
    sA0 = WMMA_BF16(qA0, k00, sA0); sA0 = WMMA_BF16(qA1, k01, sA0);
    sA1 = WMMA_BF16(qA0, k10, sA1); sA1 = WMMA_BF16(qA1, k11, sA1);
    sB0 = WMMA_BF16(qB0, k00, sB0); sB0 = WMMA_BF16(qB1, k01, sB0);
    sB1 = WMMA_BF16(qB0, k10, sB1); sB1 = WMMA_BF16(qB1, k11, sB1);

    const float scale = 0.125f;              // dh=64 -> 1/sqrt(64)
    auto soften = [&](v8f& s0, v8f& s1, float* m, float* l, v8f* oacc,
                      int qbase, int rowbase) {
      float t0[8], t1[8], corr[8];
#pragma unroll
      for (int i = 0; i < 8; i++) { t0[i] = s0[i] * scale; t1[i] = s1[i] * scale; }
      if (do_mask) {
        const int c0 = kb + ln, c1 = kb + 16 + ln;
#pragma unroll
        for (int i = 0; i < 8; i++) {
          const int q = qbase + i + 8 * hi;
          t0[i] = (c0 <= q) ? t0[i] : -1.0e30f;
          t1[i] = (c1 <= q) ? t1[i] : -1.0e30f;
        }
      }
#pragma unroll
      for (int i = 0; i < 8; i++) {
        float mt   = redmax16(fmaxf(t0[i], t1[i]));
        float mnew = fmaxf(m[i], mt);
        float c    = __expf(m[i] - mnew);
        float p0   = __expf(t0[i] - mnew);
        float p1   = __expf(t1[i] - mnew);
        l[i]       = l[i] * c + redsum16(p0 + p1);
        m[i]       = mnew;
        corr[i]    = c;
        const int row = rowbase + i + 8 * hi;
        pl[row * 32 + ln]      = f2bf(p0);
        pl[row * 32 + 16 + ln] = f2bf(p1);
      }
#pragma unroll
      for (int i = 0; i < 8; i++) {
        oacc[0][i] *= corr[i]; oacc[1][i] *= corr[i];
        oacc[2][i] *= corr[i]; oacc[3][i] *= corr[i];
      }
    };
    soften(sA0, sA1, mA, lA, oA, q0,      0);
    soften(sB0, sB1, mB, lB, oB, q0 + 16, 16);

    // V as B-operand from Vt[h][d][n]: 16 contiguous keys per lane.
    // Hoist all four V operands so their loadcnt waits overlap the PV WMMAs.
    const unsigned short* vb = Vt + (size_t)h * 64 * N + kb + 16 * hi;
    v16bf vv[4];
#pragma unroll
    for (int td = 0; td < 4; td++) {
      const unsigned short* vr = vb + (size_t)(16 * td + ln) * N;
      __builtin_prefetch(vr + 32, 0, 0);     // next chunk's V segment
      vv[td] = ld16(vr);
    }

    // Same-wave LDS write->read: C-layout -> A-layout shuffle through LDS.
    __asm__ volatile("s_wait_dscnt 0" ::: "memory");
    const v16bf paA = ld16(pl + ln * 32 + 16 * hi);
    const v16bf paB = ld16(pl + (16 + ln) * 32 + 16 * hi);
#pragma unroll
    for (int td = 0; td < 4; td++) {
      oA[td] = WMMA_BF16(paA, vv[td], oA[td]);
      oB[td] = WMMA_BF16(paB, vv[td], oB[td]);
    }
  };

  // Loop 1: always-allowed prefix, no mask.
  for (int kb = 0; kb < s_end; kb += 32) chunk(kb, false);
  // Loop 2: sliding window; causal mask only on diagonal chunks.
  int fq = q0 / fs;
  int wst = fq - w + 1; if (wst < 0) wst = 0;
  wst *= fs; if (wst < s_end) wst = s_end;
  for (int kb = wst; kb < q0 + 32; kb += 32) chunk(kb, kb + 31 > q0);

  // Normalize and store bf16 into OB[n][512] at column h*64 + d.
#pragma unroll
  for (int i = 0; i < 8; i++) {
    const float iA = 1.0f / lA[i];
    const float iB = 1.0f / lB[i];
    unsigned short* rowA = OB + (size_t)(q0 + i + 8 * hi) * 512 + h * 64;
    unsigned short* rowB = rowA + (size_t)16 * 512;
#pragma unroll
    for (int td = 0; td < 4; td++) {
      rowA[16 * td + ln] = f2bf(oA[td][i] * iA);
      rowB[16 * td + ln] = f2bf(oB[td][i] * iB);
    }
  }
}

// ---------------------------------------------------------------------------
// Output projection: out[n][co] = sum_c ob[n][c] * proj_w[co][c] + b[co]
// One wave per 16x64 strip, software-pipelined k-loop, fp32 output.
// ---------------------------------------------------------------------------
__global__ __launch_bounds__(128) void proj_gemm(
    const unsigned short* __restrict__ ob, const unsigned short* __restrict__ wb,
    const float* __restrict__ bias, float* __restrict__ out, int N) {
  const int lane = threadIdx.x & 31;
  const int wv   = threadIdx.x >> 5;
  const int ln   = lane & 15, hi = lane >> 4;
  const int TO   = 8;                        // 512/64 strips
  int tile = blockIdx.x * 4 + wv;
  int tm = tile / TO, to = tile % TO;
  if (tm * 16 >= N) return;

  const unsigned short* arow = ob + (size_t)(tm * 16 + ln) * 512;
  const unsigned short* br[4];
#pragma unroll
  for (int t = 0; t < 4; t++)
    br[t] = wb + (size_t)(to * 64 + 16 * t + ln) * 512;

  v16bf A = ld16(arow + 16 * hi);
  v16bf B[4];
#pragma unroll
  for (int t = 0; t < 4; t++) B[t] = ld16(br[t] + 16 * hi);

  v8f acc[4] = {};
#pragma unroll 2
  for (int k0 = 0; k0 < 512; k0 += 32) {
    const int kn = (k0 + 32 < 512) ? (k0 + 32) : k0;
    __builtin_prefetch(arow + k0 + 96, 0, 0);
    v16bf An = ld16(arow + kn + 16 * hi);
    v16bf Bn[4];
#pragma unroll
    for (int t = 0; t < 4; t++) {
      __builtin_prefetch(br[t] + k0 + 96, 0, 0);
      Bn[t] = ld16(br[t] + kn + 16 * hi);
    }
#pragma unroll
    for (int t = 0; t < 4; t++) acc[t] = WMMA_BF16(A, B[t], acc[t]);
    A = An;
#pragma unroll
    for (int t = 0; t < 4; t++) B[t] = Bn[t];
  }
#pragma unroll
  for (int t = 0; t < 4; t++) {
    const int co = to * 64 + 16 * t + ln;
    const float bv = bias[co];
#pragma unroll
    for (int i = 0; i < 8; i++)
      out[(size_t)(tm * 16 + i + 8 * hi) * 512 + co] = acc[t][i] + bv;
  }
}

// ---------------------------------------------------------------------------
extern "C" void kernel_launch(void* const* d_in, const int* in_sizes, int n_in,
                              void* d_out, int out_size, void* d_ws, size_t ws_size,
                              hipStream_t stream) {
  const float* x      = (const float*)d_in[0];
  const float* qkv_w  = (const float*)d_in[3];
  const float* qkv_b  = (const float*)d_in[4];
  const float* proj_w = (const float*)d_in[5];
  const float* proj_b = (const float*)d_in[6];
  const int*   p_fs   = (const int*)d_in[7];
  const int*   p_sw   = (const int*)d_in[8];
  const int*   p_nfpb = (const int*)d_in[9];
  const int*   p_nffs = (const int*)d_in[10];

  const int C = 512, H = 8;
  const int N = in_sizes[0] / C;             // B=1

  // Workspace layout (ushort elements), each region 16B-aligned.
  unsigned short* ws = (unsigned short*)d_ws;
  size_t off = 0;
  unsigned short* xb  = ws + off; off += (size_t)N * C;           // x bf16
  unsigned short* qwb = ws + off; off += (size_t)3 * C * C;       // qkv_w bf16
  unsigned short* pwb = ws + off; off += (size_t)C * C;           // proj_w bf16
  unsigned short* Qb  = ws + off; off += (size_t)H * N * 64 + 4096;
  unsigned short* Kb  = ws + off; off += (size_t)H * N * 64 + 4096;
  unsigned short* Vt  = ws + off; off += (size_t)H * 64 * N + 4096;
  unsigned short* OB  = ws + off; off += (size_t)N * C;           // attn out bf16

  {
    int n0 = N * C, n1 = 3 * C * C, n2 = C * C;
    cvt_f32_bf16<<<(n0 + 255) / 256, 256, 0, stream>>>(x, xb, n0);
    cvt_f32_bf16<<<(n1 + 255) / 256, 256, 0, stream>>>(qkv_w, qwb, n1);
    cvt_f32_bf16<<<(n2 + 255) / 256, 256, 0, stream>>>(proj_w, pwb, n2);
  }
  {
    int tiles = (N / 16) * 24;               // 16x64 strips of the QKV output
    qkv_gemm<<<(tiles + 3) / 4, 128, 0, stream>>>(xb, qwb, qkv_b, Qb, Kb, Vt, N);
  }
  {
    int tiles = H * (N / 32);                // 32-query groups
    attn_kernel<<<(tiles + 3) / 4, 128, 0, stream>>>(Qb, Kb, Vt, OB, N,
                                                     p_fs, p_sw, p_nfpb, p_nffs);
  }
  {
    int tiles = (N / 16) * 8;                // 16x64 strips of the projection
    proj_gemm<<<(tiles + 3) / 4, 128, 0, stream>>>(OB, pwb, proj_b,
                                                   (float*)d_out, N);
  }
}
